// Pofusion_ONNX_10883447128782
// MI455X (gfx1250) — 8-bit matrix operand order repaired and checked by reading, compiled, not run on hardware
//
#include <hip/hip_runtime.h>
#include <hip/hip_bf16.h>
#include <math.h>

#define B_    2
#define NS_   6
#define N_    32
#define H_    768
#define W_    768
#define HW_   (H_*W_)
#define NPIX_ (B_*HW_)
#define NEG_  (-100.0f)

#define WS_PARAMS 0
#define WS_HIST   1024
#define WS_STUFF  2048
#define WS_LUT    2080
#define WS_PRED   2176

typedef int      v8i __attribute__((ext_vector_type(8)));
typedef int      v4s __attribute__((ext_vector_type(4)));
typedef int      v8s __attribute__((ext_vector_type(8)));
typedef unsigned v4u __attribute__((ext_vector_type(4)));

#ifndef __has_builtin
#define __has_builtin(x) 0
#endif
#if __has_builtin(__builtin_amdgcn_tensor_load_to_lds) && __has_builtin(__builtin_amdgcn_s_wait_tensorcnt)
#define HAVE_TDM 1
#else
#define HAVE_TDM 0
#endif

__device__ __forceinline__ unsigned eqb(unsigned x) {
  unsigned y = (x & 0x7F7F7F7Fu) + 0x7F7F7F7Fu;
  y = ~(y | x | 0x7F7F7F7Fu);
  return y >> 7;
}

__global__ __launch_bounds__(64) void pf_pass0(const float* __restrict__ bbx,
                                               const int* __restrict__ cls,
                                               int* __restrict__ ws) {
  const int t = threadIdx.x;
  int* hist = ws + WS_HIST;
#pragma unroll
  for (int j = 0; j < 16; ++j) hist[t*16 + j] = 0;
  if (t < 16) ws[WS_STUFF + t] = 0;

  const int b = t >> 5, n = t & 31;
  const float* bb = bbx + (b*N_ + n)*4;
  const float b0 = bb[0], b1 = bb[1], b2 = bb[2], b3 = bb[3];
  const int y0 = (int)floorf(b0), x0 = (int)floorf(b1);
  const int y1 = (int)floorf(b2), x1 = (int)floorf(b3);
  const int y1r = (int)(rintf(b2) + 1.0f), x1r = (int)(rintf(b3) + 1.0f);
  const float hh = fmaxf((float)(y1 - y0 + 1), 1.0f);
  const float wwf = fmaxf((float)(x1 - x0 + 1), 1.0f);
  const int ys = max(y0, 0), ye = min(y1 + 1, H_);
  const int xs = max(x0, 0), xe = min(x1 + 1, W_);
  const int cl = cls[b*N_ + n];

  int* P = ws + WS_PARAMS + (b*N_ + n)*16;
  P[0] = y0;  P[1] = x0;  P[2] = y1r; P[3] = x1r;
  P[4] = ys;  P[5] = ye;  P[6] = xs;  P[7] = xe;
  P[8] = cl;  P[9] = ((b*N_ + n)*4 + cl)*784;
  P[10] = __float_as_int((float)y0);  P[11] = __float_as_int((float)x0);
  P[12] = __float_as_int(28.0f/hh);   P[13] = __float_as_int(28.0f/wwf);
  P[14] = 0; P[15] = 0;
}

__global__ __launch_bounds__(256) void pf_pass1(const float* __restrict__ sem,
                                                const float* __restrict__ msk,
                                                const int* __restrict__ wsp,
                                                int* __restrict__ hist,
                                                int* __restrict__ stuffp,
                                                unsigned char* __restrict__ pred8) {
  __shared__ int      s_ip[512];
  __shared__ unsigned s_k[8*32];
  __shared__ unsigned s_c[8*32];
  __shared__ int      s_bins[NS_];

  const int tid  = threadIdx.x;
  const int b    = blockIdx.x / H_;
  const int y    = blockIdx.x % H_;
  const int w    = tid >> 5, lane = tid & 31;

#if HAVE_TDM
  if (tid == 0) {
    const unsigned ldsoff = (unsigned)(unsigned long long)(const void*)s_ip;
    const unsigned long long ga = (unsigned long long)(const void*)(wsp + b*512);
    v4u g0; g0[0] = 1u;
    g0[1] = ldsoff;
    g0[2] = (unsigned)(ga & 0xFFFFFFFFull);
    g0[3] = (unsigned)((ga >> 32) & 0x01FFFFFFull) | 0x80000000u;
    v8s g1 = {0x20000,
              (512 << 16),
              (1 << 16),
              (512 << 16),
              1,
              512, 0, 0};
    v4s z4 = {0, 0, 0, 0};
#if defined(__clang_major__) && (__clang_major__ >= 23)
    v8s z8 = {0, 0, 0, 0, 0, 0, 0, 0};
    __builtin_amdgcn_tensor_load_to_lds(g0, g1, z4, z4, z8, 0);
#else
    __builtin_amdgcn_tensor_load_to_lds(g0, g1, z4, z4, 0);
#endif
    __builtin_amdgcn_s_wait_tensorcnt(0);
  }
#else
  s_ip[tid]       = wsp[b*512 + tid];
  s_ip[tid + 256] = wsp[b*512 + 256 + tid];
#endif
  if (lane < 8) { s_k[w*32 + 24 + lane] = 0xFFFFFFFFu; s_c[w*32 + 24 + lane] = 0u; }
  if (tid < NS_) s_bins[tid] = 0;
  __syncthreads();

  unsigned char* kb8 = (unsigned char*)s_k;
  unsigned char* cb8 = (unsigned char*)s_c;

  for (int i = 0; i < 3; ++i) {
    const int x = i*256 + tid;
    const float* sp0 = sem + ((size_t)(b*10)*H_ + y)*W_ + x;
    float semv[10];
#pragma unroll
    for (int c = 0; c < 10; ++c) semv[c] = sp0[(size_t)c * HW_];
    if (i < 2) __builtin_prefetch(sp0 + 256, 0, 1);

    int semp = 0; float smx = semv[0];
#pragma unroll
    for (int c = 1; c < 10; ++c) if (semv[c] > smx) { smx = semv[c]; semp = c; }

    float best = semv[0]; int bc = 0;
#pragma unroll
    for (int c = 1; c < NS_; ++c) if (semv[c] > best) { best = semv[c]; bc = c; }

    for (int n = 0; n < N_; ++n) {
      const int* P = &s_ip[n*16];
      const bool ins = (y >= P[0]) & (y < P[2]) & (x >= P[1]) & (x < P[3]);
      const bool inm = (y >= P[4]) & (y < P[5]) & (x >= P[6]) & (x < P[7]);
      float v = -0.0f;
      if (ins | inm) {
        const int cl = P[8];
        const float tl = (cl == 0) ? semv[6] : (cl == 1) ? semv[7]
                         : (cl == 2) ? semv[8] : semv[9];
        const float inst = ins ? tl : NEG_;
        float mcv = NEG_;
        if (inm) {
          float syf = ((float)y - __int_as_float(P[10]) + 0.5f) * __int_as_float(P[12]) - 0.5f;
          syf = fminf(fmaxf(syf, 0.0f), 27.0f);
          const int iy0 = (int)syf; const float wy = syf - (float)iy0;
          const int iy1 = min(iy0 + 1, 27);
          float sxf = ((float)x - __int_as_float(P[11]) + 0.5f) * __int_as_float(P[13]) - 0.5f;
          sxf = fminf(fmaxf(sxf, 0.0f), 27.0f);
          const int ix0 = (int)sxf; const float wx = sxf - (float)ix0;
          const int ix1 = min(ix0 + 1, 27);
          const float* mp = msk + P[9];
          const float m00 = mp[iy0*28 + ix0], m01 = mp[iy0*28 + ix1];
          const float m10 = mp[iy1*28 + ix0], m11 = mp[iy1*28 + ix1];
          const float ta = m00 + (m10 - m00)*wy;
          const float tb = m01 + (m11 - m01)*wy;
          mcv = ta + (tb - ta)*wx;
        }
        const float s1 = 1.0f / (1.0f + __expf(-inst));
        const float s2 = 1.0f / (1.0f + __expf(-mcv));
        v = (s1 + s2) * (inst + mcv);
      }
      if (v > best) { best = v; bc = NS_ + n; }
    }

    pred8[(size_t)(b*H_ + y)*W_ + x] = (unsigned char)bc;
    if (bc < NS_) atomicAdd(&s_bins[bc], 1);
    const int e = i*32 + lane;
    kb8[w*128 + e] = (bc >= NS_) ? (unsigned char)(bc - NS_) : (unsigned char)0xFF;
    cb8[w*128 + e] = (unsigned char)semp;
  }
  __syncthreads();

  const unsigned mrow = lane & 15u;
  const unsigned half = lane >> 4;
  const unsigned mA0 = mrow * 0x01010101u;
  const unsigned mA1 = (mrow + 16u) * 0x01010101u;
  const unsigned mB  = mrow * 0x01010101u;
  v8i acc0 = {0,0,0,0,0,0,0,0};
  v8i acc1 = {0,0,0,0,0,0,0,0};
#pragma unroll
  for (int r = 0; r < 2; ++r) {
    const unsigned* kd = &s_k[w*32 + r*16];
    const unsigned* cd = &s_c[w*32 + r*16];
    v8i a0, a1, bmat;
#pragma unroll
    for (int j = 0; j < 8; ++j) {
      const int ka = (((j >> 1)*16 + (int)half*8 + (j & 1)*4) >> 2);
      const unsigned kv = kd[ka];
      a0[j] = (int)eqb(kv ^ mA0);
      a1[j] = (int)eqb(kv ^ mA1);
      const int kb = (((j >> 1)*16 + (int)half*8 + (j & 1)*4) >> 2);
      bmat[j] = (int)eqb(cd[kb] ^ mB);
    }
    acc0 = __builtin_amdgcn_wmma_i32_16x16x64_iu8(false, a0, false, bmat, acc0, false, false);
    acc1 = __builtin_amdgcn_wmma_i32_16x16x64_iu8(false, a1, false, bmat, acc1, false, false);
  }
  const int col = (int)(lane & 15u);
  if (col < 10) {
    const int rbase = (int)half * 8;
#pragma unroll
    for (int j = 0; j < 8; ++j) {
      const int vA = acc0[j];
      if (vA) atomicAdd(&hist[(b*N_ + rbase + j)*16 + col], vA);
      const int vB = acc1[j];
      if (vB) atomicAdd(&hist[(b*N_ + 16 + rbase + j)*16 + col], vB);
    }
  }
  if (tid < NS_) { const int sv = s_bins[tid]; if (sv) atomicAdd(&stuffp[b*8 + tid], sv); }
}

__global__ __launch_bounds__(64) void pf_pass2(const int* __restrict__ ws_hist,
                                               const int* __restrict__ ws_stuff,
                                               const int* __restrict__ cls,
                                               int* __restrict__ ws_lut,
                                               int* __restrict__ out) {
  const int b = threadIdx.x;
  if (b >= B_) return;
  int total[N_], semlab[N_];
  bool keptA[N_], br2A[N_];
  int cum = 0;
  for (int n = 0; n < N_; ++n) {
    int tot = 0, mx = -1, mxc = 0;
    for (int c = 0; c < 10; ++c) {
      const int hv = ws_hist[(b*N_ + n)*16 + c];
      tot += hv;
      if (hv > mx) { mx = hv; mxc = c; }
    }
    const bool present = tot > 0;
    const int cl6 = cls[b*N_ + n] + NS_;
    const bool br1 = (mxc == cl6);
    const bool br2 = (!br1) && (2*mx >= tot) && (mxc < NS_) && present;
    semlab[n] = br2 ? mxc : cl6;
    br2A[n] = br2;
    keptA[n] = present && !br2;
    total[n] = tot;
    cum += present ? 1 : 0;
  }
  int sp[NS_];
  for (int s = 0; s < NS_; ++s) sp[s] = ws_stuff[b*8 + s];
  for (int n = 0; n < N_; ++n) if (br2A[n]) sp[semlab[n]] += total[n];
  int srank[NS_]; int ns = 0;
  for (int s = 0; s < NS_; ++s) { srank[s] = ns; if (sp[s] > 0) ns++; }

  int lut[NS_ + N_];
  for (int s = 0; s < NS_; ++s) lut[s] = (sp[s] > 0) ? (srank[s] + 1) : 0;
  int pcls[39];
  for (int l = 0; l < 39; ++l) pcls[l] = -1;
  pcls[0] = 255;
  for (int s = 0; s < NS_; ++s) if (sp[s] > 0) pcls[1 + srank[s]] = s;
  int rk = 0;
  for (int n = 0; n < N_; ++n) {
    const int ch = NS_ + n;
    if (keptA[n])      { lut[ch] = 1 + ns + rk; pcls[1 + ns + rk] = cls[b*N_ + n] + NS_; rk++; }
    else if (br2A[n])  { lut[ch] = srank[semlab[n]] + 1; }
    else               { lut[ch] = 0; }
  }
  for (int ch = 0; ch < NS_ + N_; ++ch) ws_lut[b*40 + ch] = lut[ch];
  const int vlen = 1 + ns + rk;
  int* ocls = out + NPIX_;
  int* oisc = out + NPIX_ + B_*39;
  for (int l = 0; l < 39; ++l) {
    ocls[b*39 + l] = pcls[l];
    oisc[b*39 + l] = (l < vlen) ? 0 : -1;
  }
  out[NPIX_ + 2*B_*39 + b] = vlen;
}

__global__ __launch_bounds__(256) void pf_pass3(const unsigned* __restrict__ pred4,
                                                const int* __restrict__ lut,
                                                int* __restrict__ out) {
  const int t = blockIdx.x*256 + threadIdx.x;
  const unsigned v = pred4[t];
  const int b = (t >= (HW_/4)) ? 1 : 0;
  const int* L = lut + b*40;
  int4 o;
  o.x = L[v & 255u];
  o.y = L[(v >> 8) & 255u];
  o.z = L[(v >> 16) & 255u];
  o.w = L[v >> 24];
  ((int4*)out)[t] = o;
}

extern "C" void kernel_launch(void* const* d_in, const int* in_sizes, int n_in,
                              void* d_out, int out_size, void* d_ws, size_t ws_size,
                              hipStream_t stream) {
  (void)in_sizes; (void)n_in; (void)out_size; (void)ws_size;
  const float* sem = (const float*)d_in[0];
  const float* msk = (const float*)d_in[1];
  const float* bbx = (const float*)d_in[2];
  const int*   cls = (const int*)d_in[3];
  int* out = (int*)d_out;
  int* ws  = (int*)d_ws;

  pf_pass0<<<1, 64, 0, stream>>>(bbx, cls, ws);
  pf_pass1<<<B_*H_, 256, 0, stream>>>(sem, msk,
                                      ws + WS_PARAMS, ws + WS_HIST, ws + WS_STUFF,
                                      (unsigned char*)(ws + WS_PRED));
  pf_pass2<<<1, 64, 0, stream>>>(ws + WS_HIST, ws + WS_STUFF, cls, ws + WS_LUT, out);
  pf_pass3<<<NPIX_/1024, 256, 0, stream>>>((const unsigned*)(ws + WS_PRED), ws + WS_LUT, out);
}
